// ResNet_75600014344545
// MI455X (gfx1250) — matrix operand order repaired and checked by reading, not compiled, not run on hardware
//
#include <hip/hip_runtime.h>
#include <hip/hip_bf16.h>
#include <cstdint>
#include <cstddef>

// ---------------------------------------------------------------------------
// Types for WMMA fragments (wave32, gfx1250)
// ---------------------------------------------------------------------------
typedef _Float16 v16h __attribute__((ext_vector_type(16)));
typedef _Float16 v8h  __attribute__((ext_vector_type(8)));
typedef float    v8f  __attribute__((ext_vector_type(8)));

#define EPS 1e-5f

// Q1.7 quantization exactly as the reference: floor(x*128) clipped to
// [-128,127], then (v|8)&248 (int32 semantics), then /128.
__device__ __forceinline__ float quantize(float x) {
    int s = (int)floorf(x * 128.0f);
    s = s < -128 ? -128 : (s > 127 ? 127 : s);
    int c = (s | 8) & 248;
    return (float)c * (1.0f / 128.0f);
}

// ---------------------------------------------------------------------------
// Fold eval-mode BN into per-channel scale/bias
// ---------------------------------------------------------------------------
__global__ void fold_bn(const float* __restrict__ g, const float* __restrict__ b,
                        const float* __restrict__ m, const float* __restrict__ v,
                        float* __restrict__ scale, float* __restrict__ bias, int C) {
    int i = blockIdx.x * blockDim.x + threadIdx.x;
    if (i >= C) return;
    float inv = g[i] * rsqrtf(v[i] + EPS);
    scale[i] = inv;
    bias[i]  = b[i] - m[i] * inv;
}

// ---------------------------------------------------------------------------
// Weight transform: OIHW f32 -> [K][Cout] f16, K = (ky*ks+kx)*Cin + ci
// (ci fastest so each 32-wide K chunk is one contiguous NHWC gather)
// ---------------------------------------------------------------------------
__global__ void prep_w(const float* __restrict__ w, _Float16* __restrict__ wkn,
                       int Cout, int Cin, int ks) {
    long K = (long)ks * ks * Cin;
    long idx = (long)blockIdx.x * blockDim.x + threadIdx.x;
    if (idx >= K * Cout) return;
    int  co  = (int)(idx % Cout);
    long k   = idx / Cout;
    int  ci  = (int)(k % Cin);
    int  kxy = (int)(k / Cin);
    int  ky  = kxy / ks, kx = kxy - ky * ks;
    wkn[k * Cout + co] =
        (_Float16)w[(((long)co * Cin + ci) * ks + ky) * ks + kx];
}

// ---------------------------------------------------------------------------
// Conv1: 3x3, Cin=1, stride 1, pad 1, fused BN + ReLU + quantize.
// x: NCHW f32 (C==1).  out: NHWC f16 (128,64,64,64).
// ---------------------------------------------------------------------------
__global__ void conv1_k(const float* __restrict__ x, const float* __restrict__ w,
                        const float* __restrict__ scale, const float* __restrict__ bias,
                        _Float16* __restrict__ out) {
    int p = blockIdx.x * blockDim.x + threadIdx.x;   // (n, y, x)
    if (p >= 128 * 64 * 64) return;
    int n  = p >> 12;
    int y  = (p >> 6) & 63;
    int xx = p & 63;
    float in9[9];
#pragma unroll
    for (int ky = 0; ky < 3; ++ky)
#pragma unroll
        for (int kx = 0; kx < 3; ++kx) {
            int iy = y + ky - 1, ix = xx + kx - 1;
            in9[ky * 3 + kx] = (iy >= 0 && iy < 64 && ix >= 0 && ix < 64)
                                   ? x[((long)n * 64 + iy) * 64 + ix] : 0.0f;
        }
    for (int co = 0; co < 64; ++co) {
        float acc = 0.0f;
#pragma unroll
        for (int j = 0; j < 9; ++j) acc = fmaf(in9[j], w[co * 9 + j], acc);
        float val = fmaxf(acc * scale[co] + bias[co], 0.0f);
        out[(long)p * 64 + co] = (_Float16)quantize(val);
    }
}

// ---------------------------------------------------------------------------
// Implicit-GEMM conv via V_WMMA_F32_16X16X32_F16, double-buffered LDS.
//   M = 128*Ho*Wo (output pixels), N = Cout, K = ks*ks*Cin
// Block tile 256x64, BK=32.  8 waves stacked in M; each wave owns a 32x64
// strip = 2x4 fragments = 8 WMMAs per K chunk.  The K loop is nested
// (ky, kx, ciChunk) so the hot loop has no integer division; within each
// (ky,kx) the ciChunk loop is software-pipelined: chunk i+1 is fetched from
// global into registers and stored to the alternate LDS buffer while chunk i
// is consumed by WMMAs -- one barrier per chunk.
// Epilogue: y = acc*scale + bias (+residual) (ReLU) -> f16 NHWC.
// ---------------------------------------------------------------------------
__global__ __launch_bounds__(256)
void igemm_wmma(const _Float16* __restrict__ in, const _Float16* __restrict__ wkn,
                const float* __restrict__ scale, const float* __restrict__ bias,
                const _Float16* __restrict__ resid, _Float16* __restrict__ out,
                int H, int W, int Cin, int Ho, int Wo, int Cout,
                int stride, int ks, int pad, int do_relu) {
    __shared__ __align__(128) _Float16 As[2][256][32];  // [buf][m][k]  32 KB
    __shared__ __align__(128) _Float16 Bs[2][64][32];   // [buf][n][k]   8 KB

    const int t    = threadIdx.x;
    const int lane = t & 31;
    const int wave = t >> 5;
    const int wm   = wave * 32;         // wave M offset in tile (8 waves x 32)
    const int m0   = blockIdx.x * 256;
    const int n0   = blockIdx.y * 64;
    const int HoWo = Ho * Wo;
    const int ciChunks = Cin >> 5;

    v8f acc[2][4] = {};

    // A-stage coordinates: one thread per tile row, 32 halves (64 B) each.
    const int mg    = m0 + t;
    const int n_img = mg / HoWo;
    const int rem   = mg - n_img * HoWo;
    const int oy    = rem / Wo;
    const int ox    = rem - oy * Wo;
    const int iy0   = oy * stride - pad;
    const int ix0   = ox * stride - pad;

    const int rA   = lane & 15;
    const int half = lane >> 4;

    // B-stage coordinates: 32 k-rows x 64 n-cols, 8 halves per thread.
    const int krow = t >> 3;
    const int nc0  = (t & 7) * 8;

    const _Float16* bptr = wkn + (long)krow * Cout + n0 + nc0;
    const long bstep = (long)32 * Cout;          // halves per K chunk of weights

    int pbuf = 0;
    v16h ra0, ra1;   // staged A (32 ci)
    v8h  rb;         // staged B (8 n)

    for (int ky = 0; ky < ks; ++ky) {
        const int iy  = iy0 + ky;
        const bool yok = (iy >= 0) & (iy < H);
        for (int kx = 0; kx < ks; ++kx) {
            const int ix  = ix0 + kx;
            const bool ok = yok & (ix >= 0) & (ix < W);
            const _Float16* asrc =
                ok ? in + (((long)n_img * H + iy) * W + ix) * Cin : nullptr;

            // global -> registers for chunk `cic`
            auto load_glb = [&](int cic) {
#pragma unroll
                for (int e = 0; e < 16; ++e) { ra0[e] = (_Float16)0.0f; ra1[e] = (_Float16)0.0f; }
                if (ok) {
                    const _Float16* s = asrc + (cic << 5);
                    ra0 = *(const v16h*)s;
                    ra1 = *(const v16h*)(s + 16);
                }
                rb = *(const v8h*)bptr;
                __builtin_prefetch(bptr + bstep, 0, 3);   // global_prefetch_b8
                bptr += bstep;
            };
            // registers -> LDS buffer `buf`
            auto store_lds = [&](int buf) {
                *(v16h*)&As[buf][t][0]  = ra0;
                *(v16h*)&As[buf][t][16] = ra1;
#pragma unroll
                for (int j = 0; j < 8; ++j) Bs[buf][nc0 + j][krow] = rb[j];
            };

            // ---- pipeline prologue for this (ky,kx) ----
            load_glb(0);
            store_lds(pbuf);
            __syncthreads();

            for (int cic = 0; cic < ciChunks; ++cic) {
                const bool more = (cic + 1 < ciChunks);
                if (more) load_glb(cic + 1);

                // ---- compute chunk `cic` from LDS[pbuf] ----
                // A (16x32 f16): lane(0-15) M=lane,   e0-7:K=0-7,  e8-15:K=16-23
                //                lane(16-31) M=ln-16, e0-7:K=8-15, e8-15:K=24-31
                // B (32x16 f16): lane(0-15) N=lane, e0-7:K=0-7, e8-15:K=16-23; lane(16-31) N=ln-16, e0-7:K=8-15, e8-15:K=24-31
                const _Float16 (*Ab)[32] = As[pbuf];
                const _Float16 (*Bb)[32] = Bs[pbuf];
                v16h a[2], b[4];
#pragma unroll
                for (int i = 0; i < 2; ++i) {
                    const _Float16* ap = &Ab[wm + i * 16 + rA][0];
                    v8h lo = *(const v8h*)(ap + 8 * half);
                    v8h hi = *(const v8h*)(ap + 16 + 8 * half);
                    a[i] = __builtin_shufflevector(lo, hi, 0, 1, 2, 3, 4, 5, 6, 7,
                                                           8, 9, 10, 11, 12, 13, 14, 15);
                }
#pragma unroll
                for (int i = 0; i < 4; ++i) {
                    const _Float16* bp = &Bb[i * 16 + rA][0];
                    v8h blo = *(const v8h*)(bp + 8 * half);
                    v8h bhi = *(const v8h*)(bp + 16 + 8 * half);
                    b[i] = __builtin_shufflevector(blo, bhi, 0, 1, 2, 3, 4, 5, 6, 7,
                                                           8, 9, 10, 11, 12, 13, 14, 15);
                }
#pragma unroll
                for (int mi = 0; mi < 2; ++mi)
#pragma unroll
                    for (int ni = 0; ni < 4; ++ni)
                        acc[mi][ni] = __builtin_amdgcn_wmma_f32_16x16x32_f16(
                            false, a[mi], false, b[ni], (short)0, acc[mi][ni],
                            false, false);

                // ---- stage chunk cic+1 into the other buffer ----
                if (more) store_lds(pbuf ^ 1);
                __syncthreads();
                pbuf ^= 1;
            }
        }
    }

    // ---- epilogue: scale/bias (+residual) (ReLU) -> f16 NHWC ----
#pragma unroll
    for (int mi = 0; mi < 2; ++mi)
#pragma unroll
        for (int ni = 0; ni < 4; ++ni) {
            const int co = n0 + ni * 16 + rA;
            const float sc = scale[co];
            const float bi = bias[co];
#pragma unroll
            for (int r = 0; r < 8; ++r) {
                const int  m    = m0 + wm + mi * 16 + half * 8 + r;  // C/D: M=r+8*half
                const long oidx = (long)m * Cout + co;
                float v = acc[mi][ni][r] * sc + bi;
                if (resid) v += (float)resid[oidx];
                if (do_relu) v = fmaxf(v, 0.0f);
                out[oidx] = (_Float16)v;
            }
        }
}

// ---------------------------------------------------------------------------
// Global average pool (8x8) + quantize + FC(512->10) + bias -> f32 out
// One block per image.
// ---------------------------------------------------------------------------
__global__ void pool_fc(const _Float16* __restrict__ act,
                        const float* __restrict__ fcw, const float* __restrict__ fcb,
                        float* __restrict__ out) {
    __shared__ float q[512];
    const int n = blockIdx.x;
    const int t = threadIdx.x;  // 256 threads
    for (int c = t; c < 512; c += 256) {
        float s = 0.0f;
        for (int p = 0; p < 64; ++p)
            s += (float)act[((long)n * 64 + p) * 512 + c];
        q[c] = quantize(s * (1.0f / 64.0f));
    }
    __syncthreads();
    if (t < 10) {
        float s = fcb[t];
        for (int c = 0; c < 512; ++c) s = fmaf(q[c], fcw[t * 512 + c], s);
        out[n * 10 + t] = s;
    }
}

// ---------------------------------------------------------------------------
// Host orchestration
// ---------------------------------------------------------------------------
extern "C" void kernel_launch(void* const* d_in, const int* in_sizes, int n_in,
                              void* d_out, int out_size, void* d_ws, size_t ws_size,
                              hipStream_t stream) {
    (void)in_sizes; (void)n_in; (void)out_size; (void)ws_size;

    // ---- input enumeration (setup_inputs() insertion order, depth-first) ----
    int idx = 0;
    auto nextp = [&]() { return (const float*)d_in[idx++]; };
    struct Bn { const float *g, *b, *m, *v; };
    auto nextBn = [&]() { Bn r; r.g = nextp(); r.b = nextp(); r.m = nextp(); r.v = nextp(); return r; };

    const float* x    = nextp();      // (128,1,64,64) f32
    const float* c1w  = nextp();      // (64,1,3,3)
    Bn           bn1  = nextBn();

    // ---- workspace carve-up (deterministic, identical every call) ----
    char*  ws  = (char*)d_ws;
    size_t off = 0;
    auto alloc = [&](size_t bytes) -> char* {
        off = (off + 255) & ~(size_t)255;
        char* p = ws + off;
        off += bytes;
        return p;
    };
    const size_t ACT_BYTES = (size_t)128 * 64 * 64 * 64 * 2;  // 64 MB, largest tensor
    char* bufs[3];
    bufs[0] = alloc(ACT_BYTES);
    bufs[1] = alloc(ACT_BYTES);
    bufs[2] = alloc(ACT_BYTES);
    auto other = [&](void* a, void* b) -> _Float16* {
        for (int i = 0; i < 3; ++i)
            if (bufs[i] != (char*)a && bufs[i] != (char*)b) return (_Float16*)bufs[i];
        return (_Float16*)bufs[0];
    };

    // ---- generic conv layer (fold BN, transform weights, implicit GEMM) ----
    auto run_conv = [&](const _Float16* in, int H, int W, int Cin,
                        const float* w, Bn bn, int Cout, int ks, int stride, int pad,
                        const _Float16* resid, int relu, _Float16* out, int Ho, int Wo) {
        long K = (long)ks * ks * Cin;
        _Float16* wkn = (_Float16*)alloc((size_t)K * Cout * 2);
        float* sc = (float*)alloc((size_t)Cout * 4);
        float* bi = (float*)alloc((size_t)Cout * 4);
        fold_bn<<<(Cout + 63) / 64, 64, 0, stream>>>(bn.g, bn.b, bn.m, bn.v, sc, bi, Cout);
        prep_w<<<(unsigned)((K * Cout + 255) / 256), 256, 0, stream>>>(w, wkn, Cout, Cin, ks);
        int M = 128 * Ho * Wo;
        igemm_wmma<<<dim3(M / 256, Cout / 64), 256, 0, stream>>>(
            in, wkn, sc, bi, resid, out, H, W, Cin, Ho, Wo, Cout, stride, ks, pad, relu);
    };

    // ---- conv1 + BN + ReLU + quantize -> f16 NHWC in buf0 ----
    {
        float* sc = (float*)alloc(64 * 4);
        float* bi = (float*)alloc(64 * 4);
        fold_bn<<<1, 64, 0, stream>>>(bn1.g, bn1.b, bn1.m, bn1.v, sc, bi, 64);
        conv1_k<<<(128 * 64 * 64 + 255) / 256, 256, 0, stream>>>(
            x, c1w, sc, bi, (_Float16*)bufs[0]);
    }

    // ---- residual stages ----
    const int planes_arr[4]  = {64, 128, 256, 512};
    const int strides_arr[4] = {1, 2, 2, 2};
    _Float16* cur = (_Float16*)bufs[0];
    int sp = 64, inpl = 64;

    for (int s = 0; s < 4; ++s) {
        for (int blk = 0; blk < 2; ++blk) {
            const int stride = (blk == 0) ? strides_arr[s] : 1;
            const int planes = planes_arr[s];
            const bool has_sc = (stride != 1) || (inpl != planes);

            const float* w1 = nextp(); Bn b1 = nextBn();
            const float* w2 = nextp(); Bn b2 = nextBn();
            const float* wsc = nullptr; Bn bsc{};
            if (has_sc) { wsc = nextp(); bsc = nextBn(); }

            const int Ho = sp / stride;

            // block conv1 (3x3, BN, ReLU) -> t1
            _Float16* t1 = other(cur, nullptr);
            run_conv(cur, sp, sp, inpl, w1, b1, planes, 3, stride, 1,
                     nullptr, 1, t1, Ho, Ho);

            // shortcut
            const _Float16* scp;
            _Float16* outb;
            if (has_sc) {
                _Float16* q = other(cur, t1);
                run_conv(cur, sp, sp, inpl, wsc, bsc, planes, 1, stride, 0,
                         nullptr, 0, q, Ho, Ho);
                scp  = q;
                outb = cur;              // input fully consumed; safe to overwrite
            } else {
                scp  = cur;
                outb = other(cur, t1);
            }

            // block conv2 (3x3, BN) + residual + ReLU -> outb
            run_conv(t1, Ho, Ho, planes, w2, b2, planes, 3, 1, 1,
                     scp, 1, outb, Ho, Ho);

            cur  = outb;
            sp   = Ho;
            inpl = planes;
        }
    }

    // ---- avg-pool + quantize + FC ----
    const float* fcw = nextp();   // (10,512)
    const float* fcb = nextp();   // (10,)
    pool_fc<<<128, 256, 0, stream>>>(cur, fcw, fcb, (float*)d_out);
}
